// EnhancedGNNModel_47115791237140
// MI455X (gfx1250) — matrix operand order repaired and checked by reading, not compiled, not run on hardware
//
#include <hip/hip_runtime.h>
#include <stdint.h>

#define NNODES 100000
#define NEDGES 3200000
#define HCH    128

// ---- WMMA vector types (gfx1250, wave32) ----
typedef __attribute__((ext_vector_type(16))) __bf16        v16bf;
typedef __attribute__((ext_vector_type(8)))  float         v8f;
typedef __attribute__((ext_vector_type(8)))  unsigned int  v8u;

__device__ __forceinline__ unsigned short f32_to_bf16(float f) {
    union { float f; unsigned int u; } c; c.f = f;
    unsigned int u = c.u;
    unsigned int r = u + 0x7FFFu + ((u >> 16) & 1u);   // round-to-nearest-even
    return (unsigned short)(r >> 16);
}
__device__ __forceinline__ unsigned int pack_bf16x2(float lo, float hi) {
    return (unsigned int)f32_to_bf16(lo) | ((unsigned int)f32_to_bf16(hi) << 16);
}

// ---------------- utility kernels ----------------
__global__ void zero4_kernel(float4* p, int n4) {
    int t = blockIdx.x * blockDim.x + threadIdx.x;
    if (t < n4) { float4 z; z.x = z.y = z.z = z.w = 0.0f; p[t] = z; }
}

__global__ void deg_kernel(const long long* __restrict__ dst, float* __restrict__ deg, int E) {
    int t = blockIdx.x * blockDim.x + threadIdx.x;
    if (t < E) atomicAdd(&deg[(int)dst[t]], 1.0f);
}

__global__ void invdeg_kernel(float* deg, int n) {
    int t = blockIdx.x * blockDim.x + threadIdx.x;
    if (t < n) deg[t] = 1.0f / fmaxf(deg[t], 1.0f);
}

__global__ void cvt_weights_kernel(const float* __restrict__ w0, const float* __restrict__ w1,
                                   const float* __restrict__ w2, const float* __restrict__ w3,
                                   const float* __restrict__ w4, const float* __restrict__ w5,
                                   unsigned short* __restrict__ dstw) {
    int t = blockIdx.x * blockDim.x + threadIdx.x;
    if (t >= HCH * HCH) return;
    dstw[t              ] = f32_to_bf16(w0[t]);
    dstw[t + 1 * 16384  ] = f32_to_bf16(w1[t]);
    dstw[t + 2 * 16384  ] = f32_to_bf16(w2[t]);
    dstw[t + 3 * 16384  ] = f32_to_bf16(w3[t]);
    dstw[t + 4 * 16384  ] = f32_to_bf16(w4[t]);
    dstw[t + 5 * 16384  ] = f32_to_bf16(w5[t]);
}

// ---------------- edge scatter: agg[dst] += feat[src] ----------------
// One thread per (edge, 4-float chunk): float4 gather + 4 f32 atomics (L2-resident).
__global__ void scatter_kernel(const float4* __restrict__ xin,
                               const long long* __restrict__ src,
                               const long long* __restrict__ dst,
                               float* __restrict__ agg, int E) {
    int t = blockIdx.x * blockDim.x + threadIdx.x;
    if (t >= E * 32) return;
    int e = t >> 5;
    int c = t & 31;
    int s = (int)src[e];
    int d = (int)dst[e];
    float4 v = xin[s * 32 + c];
    float* o = agg + (size_t)d * HCH + c * 4;
    atomicAdd(o + 0, v.x);
    atomicAdd(o + 1, v.y);
    atomicAdd(o + 2, v.z);
    atomicAdd(o + 3, v.w);
}

// ---------------- fused SAGE linear layer (WMMA bf16 -> f32) ----------------
// out[n,o] = relu( (agg[n,:]*inv_deg[n]) . Wl[o,:] + in[n,:] . Wr[o,:] + bias[o] )
// Block = 256 threads = 8 waves; each block handles 16 nodes x 128 outputs.
// Wave w computes output channels [16w, 16w+16) via v_wmma_f32_16x16x32_bf16,
// accumulating K=128 in 4 chunks for each of the two operand pairs.
__global__ __launch_bounds__(256)
void sage_gemm_kernel(const float* __restrict__ agg,
                      const float* __restrict__ xin,
                      const float* __restrict__ inv_deg,
                      const unsigned short* __restrict__ Wl,
                      const unsigned short* __restrict__ Wr,
                      const float* __restrict__ bias,
                      float* __restrict__ out, int relu) {
    // A operands staged in LDS as bf16 pairs, row-major [m][k/2]
    __shared__ unsigned int ldsA0[16 * 64];   // scaled aggregate
    __shared__ unsigned int ldsA1[16 * 64];   // root features

    const int tid   = threadIdx.x;
    const int node0 = blockIdx.x * 16;

    // ---- stage: each thread converts 8 consecutive floats of one row ----
    {
        int idx = tid * 8;            // 0..2047
        int m   = idx >> 7;           // row 0..15
        int k   = idx & 127;
        float invd = inv_deg[node0 + m];
        const float4* ap = (const float4*)(agg + (size_t)(node0 + m) * HCH + k);
        float4 a0 = ap[0], a1 = ap[1];
        unsigned int* dA = &ldsA0[m * 64 + (k >> 1)];
        dA[0] = pack_bf16x2(a0.x * invd, a0.y * invd);
        dA[1] = pack_bf16x2(a0.z * invd, a0.w * invd);
        dA[2] = pack_bf16x2(a1.x * invd, a1.y * invd);
        dA[3] = pack_bf16x2(a1.z * invd, a1.w * invd);
        const float4* xp = (const float4*)(xin + (size_t)(node0 + m) * HCH + k);
        float4 x0 = xp[0], x1 = xp[1];
        unsigned int* dX = &ldsA1[m * 64 + (k >> 1)];
        dX[0] = pack_bf16x2(x0.x, x0.y);
        dX[1] = pack_bf16x2(x0.z, x0.w);
        dX[2] = pack_bf16x2(x1.x, x1.y);
        dX[3] = pack_bf16x2(x1.z, x1.w);
    }
    __syncthreads();

    const int wave = tid >> 5;
    const int lane = tid & 31;
    const int n    = lane & 15;        // output column within tile / A row
    const int hi   = lane >> 4;        // lane-half selector
    const int o    = wave * 16 + n;    // output channel

    const unsigned int* wlu = (const unsigned int*)Wl;
    const unsigned int* wru = (const unsigned int*)Wr;

    v8f acc = {};

    // ---- operand pair 0: scaled aggregate x Wl ----
    #pragma unroll
    for (int kc = 0; kc < 4; ++kc) {
        const int K0 = kc * 32;
        v8u ar;
        const unsigned int* rowA = &ldsA0[n * 64];
        int abase = (K0 + hi * 8) >> 1;         // bf16-pair index
        #pragma unroll
        for (int j = 0; j < 4; ++j) ar[j] = rowA[abase + j];          // K = K0+hi*8 + 2j
        #pragma unroll
        for (int j = 0; j < 4; ++j) ar[4 + j] = rowA[abase + 8 + j];  // K = K0+16+hi*8 + 2j
        v8u br;
        int bbase = o * 64 + ((K0 + hi * 8) >> 1);
        #pragma unroll
        for (int j = 0; j < 8; ++j) br[j] = wlu[bbase + j + (j >> 2) * 4];
        acc = __builtin_amdgcn_wmma_f32_16x16x32_bf16(
                  false, __builtin_bit_cast(v16bf, ar),
                  false, __builtin_bit_cast(v16bf, br),
                  (short)0, acc, false, false);
    }
    // ---- operand pair 1: root features x Wr ----
    #pragma unroll
    for (int kc = 0; kc < 4; ++kc) {
        const int K0 = kc * 32;
        v8u ar;
        const unsigned int* rowA = &ldsA1[n * 64];
        int abase = (K0 + hi * 8) >> 1;
        #pragma unroll
        for (int j = 0; j < 4; ++j) ar[j] = rowA[abase + j];
        #pragma unroll
        for (int j = 0; j < 4; ++j) ar[4 + j] = rowA[abase + 8 + j];
        v8u br;
        int bbase = o * 64 + ((K0 + hi * 8) >> 1);
        #pragma unroll
        for (int j = 0; j < 8; ++j) br[j] = wru[bbase + j + (j >> 2) * 4];
        acc = __builtin_amdgcn_wmma_f32_16x16x32_bf16(
                  false, __builtin_bit_cast(v16bf, ar),
                  false, __builtin_bit_cast(v16bf, br),
                  (short)0, acc, false, false);
    }

    // ---- epilogue: bias + relu, scatter per D-matrix layout ----
    float bv = bias[o];
    #pragma unroll
    for (int r = 0; r < 8; ++r) {
        int m = r + hi * 8;
        float v = acc[r] + bv;
        if (relu) v = fmaxf(v, 0.0f);
        out[(size_t)(node0 + m) * HCH + o] = v;
    }
}

// ---------------- heads: 128 -> 3/2/3, scalar f32 (shared aggregate) ----------------
__global__ void head_kernel(const float* __restrict__ agg, const float* __restrict__ h,
                            const float* __restrict__ inv_deg,
                            const float* __restrict__ Wal, const float* __restrict__ War, const float* __restrict__ ba,
                            const float* __restrict__ Wsl, const float* __restrict__ Wsr, const float* __restrict__ bs,
                            const float* __restrict__ Wel, const float* __restrict__ Wer, const float* __restrict__ be,
                            float* __restrict__ out) {
    int t = blockIdx.x * blockDim.x + threadIdx.x;
    if (t >= NNODES * 8) return;
    int node = t >> 3;
    int c    = t & 7;
    const float *Wlp, *Wrp; float bv; size_t ooff; int ch;
    if (c < 3)      { ch = c;     Wlp = Wal; Wrp = War; bv = ba[ch]; ooff = (size_t)node * 3 + ch; }
    else if (c < 5) { ch = c - 3; Wlp = Wsl; Wrp = Wsr; bv = bs[ch]; ooff = (size_t)NNODES * 3 + (size_t)node * 2 + ch; }
    else            { ch = c - 5; Wlp = Wel; Wrp = Wer; bv = be[ch]; ooff = (size_t)NNODES * 5 + (size_t)node * 3 + ch; }
    float invd = inv_deg[node];
    const float* ar  = agg + (size_t)node * HCH;
    const float* hr  = h   + (size_t)node * HCH;
    const float* wlr = Wlp + ch * HCH;
    const float* wrr = Wrp + ch * HCH;
    float s = bv;
    #pragma unroll 4
    for (int k = 0; k < HCH; ++k)
        s += ar[k] * invd * wlr[k] + hr[k] * wrr[k];
    out[ooff] = s;
}

// ---------------- launch ----------------
extern "C" void kernel_launch(void* const* d_in, const int* in_sizes, int n_in,
                              void* d_out, int out_size, void* d_ws, size_t ws_size,
                              hipStream_t stream) {
    (void)in_sizes; (void)n_in; (void)out_size; (void)ws_size;

    const float*     x   = (const float*)d_in[0];
    const long long* ei  = (const long long*)d_in[1];
    const long long* src = ei;
    const long long* dst = ei + NEDGES;
    const float* W1l = (const float*)d_in[2];
    const float* W1r = (const float*)d_in[3];
    const float* b1  = (const float*)d_in[4];
    const float* W2l = (const float*)d_in[5];
    const float* W2r = (const float*)d_in[6];
    const float* b2  = (const float*)d_in[7];
    const float* W3l = (const float*)d_in[8];
    const float* W3r = (const float*)d_in[9];
    const float* b3  = (const float*)d_in[10];
    const float* Wal = (const float*)d_in[11];
    const float* War = (const float*)d_in[12];
    const float* ba  = (const float*)d_in[13];
    const float* Wsl = (const float*)d_in[14];
    const float* Wsr = (const float*)d_in[15];
    const float* bs  = (const float*)d_in[16];
    const float* Wel = (const float*)d_in[17];
    const float* Wer = (const float*)d_in[18];
    const float* be  = (const float*)d_in[19];

    // workspace layout (floats)
    const size_t NPAD = 100352;                 // deg/inv_deg region (padded, mult of 4)
    const size_t FEAT = (size_t)NNODES * HCH;   // 12.8M floats
    float* ws   = (float*)d_ws;
    float* deg  = ws;
    float* agg  = ws + NPAD;
    float* hA   = agg + FEAT;
    float* hB   = hA + FEAT;
    unsigned short* wbf = (unsigned short*)(hB + FEAT);   // 6 x 16384 bf16

    const int B = 256;
    const int AGG4  = (int)(FEAT / 4);          // 3.2M float4s
    const int SCAT  = NEDGES * 32;              // 102.4M threads

    // degrees -> inv_deg (stored in deg)
    zero4_kernel<<<(int)(NPAD / 4 + B - 1) / B, B, 0, stream>>>((float4*)deg, (int)(NPAD / 4));
    deg_kernel<<<(NEDGES + B - 1) / B, B, 0, stream>>>(dst, deg, NEDGES);
    invdeg_kernel<<<(NNODES + B - 1) / B, B, 0, stream>>>(deg, NNODES);

    // bf16 weights for the three 128x128 layers
    cvt_weights_kernel<<<(HCH * HCH + B - 1) / B, B, 0, stream>>>(W1l, W1r, W2l, W2r, W3l, W3r, wbf);

    // layer 1: x -> hA
    zero4_kernel<<<(AGG4 + B - 1) / B, B, 0, stream>>>((float4*)agg, AGG4);
    scatter_kernel<<<(SCAT + B - 1) / B, B, 0, stream>>>((const float4*)x, src, dst, agg, NEDGES);
    sage_gemm_kernel<<<NNODES / 16, B, 0, stream>>>(agg, x, deg, wbf + 0 * 16384, wbf + 1 * 16384, b1, hA, 1);

    // layer 2: hA -> hB
    zero4_kernel<<<(AGG4 + B - 1) / B, B, 0, stream>>>((float4*)agg, AGG4);
    scatter_kernel<<<(SCAT + B - 1) / B, B, 0, stream>>>((const float4*)hA, src, dst, agg, NEDGES);
    sage_gemm_kernel<<<NNODES / 16, B, 0, stream>>>(agg, hA, deg, wbf + 2 * 16384, wbf + 3 * 16384, b2, hB, 1);

    // layer 3: hB -> hA
    zero4_kernel<<<(AGG4 + B - 1) / B, B, 0, stream>>>((float4*)agg, AGG4);
    scatter_kernel<<<(SCAT + B - 1) / B, B, 0, stream>>>((const float4*)hB, src, dst, agg, NEDGES);
    sage_gemm_kernel<<<NNODES / 16, B, 0, stream>>>(agg, hB, deg, wbf + 4 * 16384, wbf + 5 * 16384, b3, hA, 1);

    // heads: single shared aggregation of hA, then three tiny linears
    zero4_kernel<<<(AGG4 + B - 1) / B, B, 0, stream>>>((float4*)agg, AGG4);
    scatter_kernel<<<(SCAT + B - 1) / B, B, 0, stream>>>((const float4*)hA, src, dst, agg, NEDGES);
    head_kernel<<<(NNODES * 8 + B - 1) / B, B, 0, stream>>>(agg, hA, deg,
                                                            Wal, War, ba, Wsl, Wsr, bs, Wel, Wer, be,
                                                            (float*)d_out);
}
